// MoERuntimeExperts_1967095021950
// MI455X (gfx1250) — matrix operand order repaired and checked by reading, not compiled, not run on hardware
//
#include <hip/hip_runtime.h>
#include <hip/hip_bf16.h>
#include <math.h>

// ---------------------------------------------------------------------------
// MoE FFN for MI455X (gfx1250, wave32).
//   layer1: h = gelu(x @ W1[e] + b1[e]);  layer2: y = h @ W2[e] + b2[e]
// Grouped GEMM over expert-bucketed tokens; v_wmma_f32_16x16x32_bf16 core,
// fp32 accumulation, vectorized staging with packed bf16 convert, hoisted
// gather pointers (no per-step perm dependency), padded LDS, weight prefetch.
// ---------------------------------------------------------------------------

typedef __attribute__((ext_vector_type(16))) __bf16 v16bf;
typedef __attribute__((ext_vector_type(8)))  __bf16 v8bf;
typedef __attribute__((ext_vector_type(4)))  __bf16 v4bf;
typedef __attribute__((ext_vector_type(8)))  float  v8f;

#define T_TOK  8192
#define DMODEL 1024
#define HDIM   4096
#define NEXP   8

#define BM 128   // token-tile rows per block
#define BN 128   // output-col tile per block
#define BK 32    // K step == WMMA bf16 K
#define BKP 40   // padded LDS row stride (80 B: 16B-aligned, bank step 20)

// ---- fragment loads matching CDNA5 WMMA VGPR layouts (CDNA5 ISA §7.12.2) --
// A tile in LDS: [BM][BKP] row-major bf16. 16-bit A 16x32:
//   lanes 0-15  (M=lane):    v0..3 = K[0..7],  v4..7 = K[16..23]
//   lanes 16-31 (M=lane-16): v0..3 = K[8..15], v4..7 = K[24..31]
__device__ __forceinline__ v16bf lds_a_frag(const __bf16* sA, int m0, int lane) {
    const int row  = m0 + (lane & 15);
    const int koff = (lane >> 4) << 3;          // 0 or 8
    const __bf16* p = sA + row * BKP + koff;
    v8bf lo = *(const v8bf*)(p);                // K[koff .. koff+7]
    v8bf hi = *(const v8bf*)(p + 16);           // K[16+koff .. 16+koff+7]
    return __builtin_shufflevector(lo, hi, 0,1,2,3,4,5,6,7,8,9,10,11,12,13,14,15);
}

// B tile in LDS stored TRANSPOSED: [BN][BKP] so each lane's K run is
// contiguous. 16-bit B 32x16:
//   lanes 0-15  (N=lane):    K[0..7],K[16..23];  lanes 16-31 (N=lane-16): K[8..15],K[24..31]
__device__ __forceinline__ v16bf lds_b_frag(const __bf16* sB, int n0, int lane) {
    const int col  = n0 + (lane & 15);
    const int koff = (lane >> 4) << 3;          // 0 or 8
    const __bf16* p = sB + col * BKP + koff;
    v8bf lo = *(const v8bf*)(p);
    v8bf hi = *(const v8bf*)(p + 16);
    return __builtin_shufflevector(lo, hi, 0,1,2,3,4,5,6,7,8,9,10,11,12,13,14,15);
}

__device__ __forceinline__ v4bf cvt4(float a, float b, float c, float d) {
    v4bf o; o[0] = (__bf16)a; o[1] = (__bf16)b; o[2] = (__bf16)c; o[3] = (__bf16)d;
    return o;
}

__device__ __forceinline__ float gelu_erf(float v) {
    return 0.5f * v * (1.0f + erff(v * 0.70710678118654752f));
}

// ---------------------------- bucketing kernels ----------------------------
__global__ void moe_init(int* counts, int* cursors) {
    int t = threadIdx.x;
    if (t < NEXP) { counts[t] = 0; cursors[t] = 0; }
}

__global__ void moe_count(const int* __restrict__ idx, int* __restrict__ counts, int n) {
    int t = blockIdx.x * blockDim.x + threadIdx.x;
    if (t < n) atomicAdd(&counts[idx[t] & (NEXP - 1)], 1);
}

__global__ void moe_prefix(const int* __restrict__ counts, int* __restrict__ offsets) {
    if (threadIdx.x == 0) {
        int s = 0;
        for (int e = 0; e < NEXP; ++e) { offsets[e] = s; s += counts[e]; }
    }
}

__global__ void moe_scatter(const int* __restrict__ idx, const int* __restrict__ offsets,
                            int* __restrict__ cursors, int* __restrict__ perm, int n) {
    int t = blockIdx.x * blockDim.x + threadIdx.x;
    if (t < n) {
        int e = idx[t] & (NEXP - 1);
        int p = atomicAdd(&cursors[e], 1);
        perm[offsets[e] + p] = t;
    }
}

// ------------------------------- GEMM layer 1 ------------------------------
// grid = (HDIM/BN, ceil(T/BM), NEXP); block = 256 (8 waves, 2x4 wave grid)
__global__ __launch_bounds__(256) void moe_gemm1(
    const float* __restrict__ x, const float* __restrict__ w1,
    const float* __restrict__ b1, const int* __restrict__ counts,
    const int* __restrict__ offsets, const int* __restrict__ perm,
    __bf16* __restrict__ hbuf)
{
    __shared__ __bf16 sA[2][BM * BKP];
    __shared__ __bf16 sB[2][BN * BKP];

    const int e     = blockIdx.z;
    const int cnt   = counts[e];
    const int mBase = blockIdx.y * BM;
    if (mBase >= cnt) return;                   // block-uniform early exit
    const int offE  = offsets[e];
    const int nBase = blockIdx.x * BN;

    const int tid  = threadIdx.x;
    const int lane = tid & 31;
    const int wave = tid >> 5;
    const int wm   = (wave & 1) * 64;           // wave sub-tile: 64 x 32
    const int wn   = (wave >> 1) * 32;

    const float* w = w1 + (size_t)e * DMODEL * HDIM;

    v8f acc[4][2];
    #pragma unroll
    for (int i = 0; i < 4; ++i)
        #pragma unroll
        for (int j = 0; j < 2; ++j) acc[i][j] = (v8f)0.0f;

    // ---- hoisted A-gather: row pointers are K-invariant; tail rows use a
    // clamped (always-valid) pointer and a cndmask zero-select (EXEC stays 1s)
    const int cA = (tid & 7) * 4;               // 8 threads per row, 4 floats
    const float* aPtr[4];
    bool aOk[4];
    #pragma unroll
    for (int p = 0; p < 4; ++p) {
        const int jrow = mBase + p * 32 + (tid >> 3);
        aOk[p] = jrow < cnt;
        const int jr = aOk[p] ? jrow : (cnt - 1);
        aPtr[p] = x + (size_t)perm[offE + jr] * DMODEL + cA;
    }
    auto stageA = [&](int kBase, __bf16* dst) {
        #pragma unroll
        for (int p = 0; p < 4; ++p) {
            const int r = p * 32 + (tid >> 3);
            float4 v = *(const float4*)(aPtr[p] + kBase);
            if (!aOk[p]) v = make_float4(0.f, 0.f, 0.f, 0.f);
            *(v4bf*)(dst + r * BKP + cA) = cvt4(v.x, v.y, v.z, v.w);
        }
    };

    // ---- B: one base pointer; all 16 k-row loads use immediate offsets
    const int nB   = tid & (BN - 1);
    const int kB   = (tid >> 7) * 4;            // 0 or 4
    const int nLds = nB * BKP + kB;
    const float* wB = w + (size_t)kB * HDIM + nBase + nB;
    auto stageB = [&](int kBase, __bf16* dst) {
        const float* g = wB + (size_t)kBase * HDIM;
        #pragma unroll
        for (int p = 0; p < 4; ++p) {
            float f0 = g[(size_t)(p * 8 + 0) * HDIM];
            float f1 = g[(size_t)(p * 8 + 1) * HDIM];
            float f2 = g[(size_t)(p * 8 + 2) * HDIM];
            float f3 = g[(size_t)(p * 8 + 3) * HDIM];
            *(v4bf*)(dst + nLds + p * 8) = cvt4(f0, f1, f2, f3);
        }
    };

    stageA(0, sA[0]);
    stageB(0, sB[0]);
    __syncthreads();

    const int nK = DMODEL / BK;                 // 32
    for (int ks = 0; ks < nK; ++ks) {
        const int cur = ks & 1;
        if (ks + 2 < nK)                        // prefetch weight stream +2 steps
            __builtin_prefetch(w + (size_t)((ks + 2) * BK + (tid & 31)) * HDIM
                                 + nBase + ((tid >> 5) << 4), 0, 1);
        if (ks + 1 < nK) {                      // double-buffer stage +1 step
            stageA((ks + 1) * BK, sA[cur ^ 1]);
            stageB((ks + 1) * BK, sB[cur ^ 1]);
        }
        v16bf af[4], bfm[2];
        #pragma unroll
        for (int i = 0; i < 4; ++i) af[i]  = lds_a_frag(sA[cur], wm + i * 16, lane);
        #pragma unroll
        for (int j = 0; j < 2; ++j) bfm[j] = lds_b_frag(sB[cur], wn + j * 16, lane);
        #pragma unroll
        for (int i = 0; i < 4; ++i)
            #pragma unroll
            for (int j = 0; j < 2; ++j)
                acc[i][j] = __builtin_amdgcn_wmma_f32_16x16x32_bf16(
                    false, af[i], false, bfm[j], (short)0, acc[i][j], false, false);
        __syncthreads();
    }

    // epilogue: + bias1, exact-erf GELU, bf16 store into sorted h buffer
    const float* bias = b1 + (size_t)e * HDIM;
    #pragma unroll
    for (int i = 0; i < 4; ++i) {
        #pragma unroll
        for (int j = 0; j < 2; ++j) {
            const int col = nBase + wn + j * 16 + (lane & 15);
            const float bv = bias[col];
            #pragma unroll
            for (int r = 0; r < 8; ++r) {
                const int rowL = wm + i * 16 + ((lane >> 4) << 3) + r;
                const int jrow = mBase + rowL;
                if (jrow < cnt) {
                    float v = acc[i][j][r] + bv;
                    hbuf[(size_t)(offE + jrow) * HDIM + col] = (__bf16)gelu_erf(v);
                }
            }
        }
    }
}

// ------------------------------- GEMM layer 2 ------------------------------
// grid = (DMODEL/BN, ceil(T/BM), NEXP); A rows are contiguous bf16 h.
__global__ __launch_bounds__(256) void moe_gemm2(
    const __bf16* __restrict__ hbuf, const float* __restrict__ w2,
    const float* __restrict__ b2, const int* __restrict__ counts,
    const int* __restrict__ offsets, const int* __restrict__ perm,
    float* __restrict__ out)
{
    __shared__ __bf16 sA[2][BM * BKP];
    __shared__ __bf16 sB[2][BN * BKP];

    const int e     = blockIdx.z;
    const int cnt   = counts[e];
    const int mBase = blockIdx.y * BM;
    if (mBase >= cnt) return;
    const int offE  = offsets[e];
    const int nBase = blockIdx.x * BN;

    const int tid  = threadIdx.x;
    const int lane = tid & 31;
    const int wave = tid >> 5;
    const int wm   = (wave & 1) * 64;
    const int wn   = (wave >> 1) * 32;

    const float* w = w2 + (size_t)e * HDIM * DMODEL;

    v8f acc[4][2];
    #pragma unroll
    for (int i = 0; i < 4; ++i)
        #pragma unroll
        for (int j = 0; j < 2; ++j) acc[i][j] = (v8f)0.0f;

    // ---- hoisted A row pointers (contiguous bf16 rows, no conversion)
    const int cA = (tid & 3) * 8;               // 4 threads per row, 8 halves
    const __bf16* aPtr[2];
    bool aOk[2];
    #pragma unroll
    for (int p = 0; p < 2; ++p) {
        const int jrow = mBase + p * 64 + (tid >> 2);
        aOk[p] = jrow < cnt;
        const int jr = aOk[p] ? jrow : (cnt - 1);
        aPtr[p] = hbuf + (size_t)(offE + jr) * HDIM + cA;
    }
    auto stageA = [&](int kBase, __bf16* dst) {
        #pragma unroll
        for (int p = 0; p < 2; ++p) {
            const int r = p * 64 + (tid >> 2);
            v8bf v = *(const v8bf*)(aPtr[p] + kBase);
            if (!aOk[p]) v = (v8bf)(__bf16)0.0f;
            *(v8bf*)(dst + r * BKP + cA) = v;
        }
    };

    const int nB   = tid & (BN - 1);
    const int kB   = (tid >> 7) * 4;
    const int nLds = nB * BKP + kB;
    const float* wB = w + (size_t)kB * DMODEL + nBase + nB;
    auto stageB = [&](int kBase, __bf16* dst) {
        const float* g = wB + (size_t)kBase * DMODEL;
        #pragma unroll
        for (int p = 0; p < 4; ++p) {
            float f0 = g[(size_t)(p * 8 + 0) * DMODEL];
            float f1 = g[(size_t)(p * 8 + 1) * DMODEL];
            float f2 = g[(size_t)(p * 8 + 2) * DMODEL];
            float f3 = g[(size_t)(p * 8 + 3) * DMODEL];
            *(v4bf*)(dst + nLds + p * 8) = cvt4(f0, f1, f2, f3);
        }
    };

    stageA(0, sA[0]);
    stageB(0, sB[0]);
    __syncthreads();

    const int nK = HDIM / BK;                   // 128
    for (int ks = 0; ks < nK; ++ks) {
        const int cur = ks & 1;
        if (ks + 2 < nK)
            __builtin_prefetch(w + (size_t)((ks + 2) * BK + (tid & 31)) * DMODEL
                                 + nBase + ((tid >> 5) << 4), 0, 1);
        if (ks + 1 < nK) {
            stageA((ks + 1) * BK, sA[cur ^ 1]);
            stageB((ks + 1) * BK, sB[cur ^ 1]);
        }
        v16bf af[4], bfm[2];
        #pragma unroll
        for (int i = 0; i < 4; ++i) af[i]  = lds_a_frag(sA[cur], wm + i * 16, lane);
        #pragma unroll
        for (int j = 0; j < 2; ++j) bfm[j] = lds_b_frag(sB[cur], wn + j * 16, lane);
        #pragma unroll
        for (int i = 0; i < 4; ++i)
            #pragma unroll
            for (int j = 0; j < 2; ++j)
                acc[i][j] = __builtin_amdgcn_wmma_f32_16x16x32_bf16(
                    false, af[i], false, bfm[j], (short)0, acc[i][j], false, false);
        __syncthreads();
    }

    // epilogue: + bias2, fp32 scatter-store to token rows
    const float* bias = b2 + (size_t)e * DMODEL;
    #pragma unroll
    for (int i = 0; i < 4; ++i) {
        #pragma unroll
        for (int j = 0; j < 2; ++j) {
            const int col = nBase + wn + j * 16 + (lane & 15);
            const float bv = bias[col];
            #pragma unroll
            for (int r = 0; r < 8; ++r) {
                const int rowL = wm + i * 16 + ((lane >> 4) << 3) + r;
                const int jrow = mBase + rowL;
                if (jrow < cnt) {
                    const int tok = perm[offE + jrow];          // scatter
                    out[(size_t)tok * DMODEL + col] = acc[i][j][r] + bv;
                }
            }
        }
    }
}

// -------------------------------- launcher ---------------------------------
extern "C" void kernel_launch(void* const* d_in, const int* in_sizes, int n_in,
                              void* d_out, int out_size, void* d_ws, size_t ws_size,
                              hipStream_t stream) {
    (void)in_sizes; (void)n_in; (void)out_size; (void)ws_size;

    const float* x  = (const float*)d_in[0];
    const int*   idx= (const int*)d_in[1];     // expert ids (masked &7 in-kernel)
    const float* w1 = (const float*)d_in[2];
    const float* w2 = (const float*)d_in[3];
    const float* b1 = (const float*)d_in[4];
    const float* b2 = (const float*)d_in[5];
    float* out = (float*)d_out;

    // workspace layout: [counts 8][offsets 8][cursors 8][pad][perm T][h bf16 T*H]
    int* counts  = (int*)d_ws;
    int* offsets = counts + 8;
    int* cursors = counts + 16;
    int* perm    = counts + 32;
    size_t hOff  = (((size_t)(32 + T_TOK) * sizeof(int)) + 255) & ~(size_t)255;
    __bf16* hbuf = (__bf16*)((char*)d_ws + hOff);

    moe_init   <<<1, 32, 0, stream>>>(counts, cursors);
    moe_count  <<<(T_TOK + 255) / 256, 256, 0, stream>>>(idx, counts, T_TOK);
    moe_prefix <<<1, 32, 0, stream>>>(counts, offsets);
    moe_scatter<<<(T_TOK + 255) / 256, 256, 0, stream>>>(idx, offsets, cursors, perm, T_TOK);

    dim3 g1(HDIM / BN, (T_TOK + BM - 1) / BM, NEXP);
    moe_gemm1<<<g1, 256, 0, stream>>>(x, w1, b1, counts, offsets, perm, hbuf);

    dim3 g2(DMODEL / BN, (T_TOK + BM - 1) / BM, NEXP);
    moe_gemm2<<<g2, 256, 0, stream>>>(hbuf, w2, b2, counts, offsets, perm, out);
}
